// Attention_16741782520042
// MI455X (gfx1250) — compile-verified, matrix operand order checked by reading, not run on hardware
//
#include <hip/hip_runtime.h>

#define S_LEN 2048
#define D_DIM 4096
#define H_NUM 32
#define HD_DIM 128

typedef __attribute__((ext_vector_type(16))) __bf16 bf16x16;
typedef __attribute__((ext_vector_type(8)))  __bf16 bf16x8;
typedef __attribute__((ext_vector_type(8)))  float  f32x8;
typedef __attribute__((ext_vector_type(4)))  int    i32x4;
typedef __attribute__((address_space(1)))    i32x4  as1_i32x4;
typedef __attribute__((address_space(3)))    i32x4  as3_i32x4;

__device__ __forceinline__ f32x8 wmma_bf16(bf16x16 a, bf16x16 b, f32x8 c) {
  return __builtin_amdgcn_wmma_f32_16x16x32_bf16(false, a, false, b, (short)0, c,
                                                 false, false);
}

// ---- async global->LDS copy (16 bytes / lane), with sync fallback ----------
#if defined(__gfx1250__) && __has_builtin(__builtin_amdgcn_global_load_async_to_lds_b128)
#define HAVE_ASYNC_LDS 1
#endif

__device__ __forceinline__ void async_copy16(const __bf16* gsrc, __bf16* ldst) {
#ifdef HAVE_ASYNC_LDS
  __builtin_amdgcn_global_load_async_to_lds_b128(
      (as1_i32x4*)(i32x4*)const_cast<__bf16*>(gsrc),
      (as3_i32x4*)(i32x4*)ldst, 0, 0);
#else
  *(bf16x8*)ldst = *(const bf16x8*)gsrc;
#endif
}
__device__ __forceinline__ void async_join() {
#ifdef HAVE_ASYNC_LDS
#if __has_builtin(__builtin_amdgcn_s_wait_asynccnt)
  __builtin_amdgcn_s_wait_asynccnt(0);
#else
  asm volatile("s_wait_asynccnt 0x0" ::: "memory");
#endif
#endif
}

// Load a 16x32 bf16 A/B fragment from a row-major tile (row stride `ld` elems).
// CDNA5 16-bit A/B layout: lanes 0-15 -> row=lane, K {0..7,16..23};
// lanes 16-31 -> row=lane-16, K {8..15,24..31}.
__device__ __forceinline__ bf16x16 load_frag(const __bf16* base, int ld) {
  const int lane = threadIdx.x & 31;
  const int r  = lane & 15;
  const int h2 = lane >> 4;
  const __bf16* p = base + (size_t)r * ld + h2 * 8;
  bf16x8 lo = *(const bf16x8*)(p);
  bf16x8 hi = *(const bf16x8*)(p + 16);
  bf16x16 f;
#pragma unroll
  for (int i = 0; i < 8; ++i) { f[i] = lo[i]; f[i + 8] = hi[i]; }
  return f;
}

__device__ __forceinline__ float rmax16(float v) {
#pragma unroll
  for (int m = 1; m <= 8; m <<= 1) v = fmaxf(v, __shfl_xor(v, m, 32));
  return v;
}
__device__ __forceinline__ float rsum16(float v) {
#pragma unroll
  for (int m = 1; m <= 8; m <<= 1) v += __shfl_xor(v, m, 32);
  return v;
}

// ---------------------------------------------------------------------------
// C(MxN,f32) = A(MxK,f32) * B(KxN,f32) using bf16 WMMA, f32 accumulate.
// 128x128 tile per 256-thread WG (8 waves), BK=32. Double-buffered LDS:
// next tile's global loads are issued before computing the current tile.
// ---------------------------------------------------------------------------
#define GLDT 40
__global__ __launch_bounds__(256) void gemm_wmma(const float* __restrict__ A,
                                                 const float* __restrict__ Bw,
                                                 float* __restrict__ C,
                                                 int M, int N, int K) {
  __shared__ __align__(16) __bf16 Asm[2][128 * GLDT];
  __shared__ __align__(16) __bf16 Bsm[2][128 * GLDT];  // transposed [n][k]
  const int tid  = threadIdx.x;
  const int wave = tid >> 5;
  const int lane = tid & 31;
  const int wr = wave >> 1;   // 0..3  -> 32-row strip
  const int wc = wave & 1;    // 0..1  -> 64-col strip
  const int m0 = blockIdx.y * 128;
  const int n0 = blockIdx.x * 128;

  f32x8 acc[2][4] = {};
  float4 va[4], vb[4];

  auto gload = [&](int k0) {  // issue 8 b128 loads, no waits in between
#pragma unroll
    for (int it = 0; it < 4; ++it) {           // A 128x32: 8 float4 per row
      int slot = it * 256 + tid;
      int row = slot >> 3, col4 = (slot & 7) * 4;
      int grow = m0 + row; if (grow >= M) grow = M - 1;
      va[it] = *(const float4*)(A + (size_t)grow * K + k0 + col4);
    }
#pragma unroll
    for (int it = 0; it < 4; ++it) {           // B 32x128: 32 float4 per k-row
      int slot = it * 256 + tid;
      int krow = slot >> 5, col4 = (slot & 31) * 4;
      vb[it] = *(const float4*)(Bw + (size_t)(k0 + krow) * N + n0 + col4);
    }
  };
  auto lstore = [&](int buf) {  // f32->bf16 convert + LDS store
#pragma unroll
    for (int it = 0; it < 4; ++it) {
      int slot = it * 256 + tid;
      int row = slot >> 3, col4 = (slot & 7) * 4;
      __bf16* dst = Asm[buf] + row * GLDT + col4;
      dst[0] = (__bf16)va[it].x; dst[1] = (__bf16)va[it].y;
      dst[2] = (__bf16)va[it].z; dst[3] = (__bf16)va[it].w;
    }
#pragma unroll
    for (int it = 0; it < 4; ++it) {
      int slot = it * 256 + tid;
      int krow = slot >> 5, col4 = (slot & 31) * 4;
      Bsm[buf][(col4 + 0) * GLDT + krow] = (__bf16)vb[it].x;
      Bsm[buf][(col4 + 1) * GLDT + krow] = (__bf16)vb[it].y;
      Bsm[buf][(col4 + 2) * GLDT + krow] = (__bf16)vb[it].z;
      Bsm[buf][(col4 + 3) * GLDT + krow] = (__bf16)vb[it].w;
    }
  };

  gload(0);
  lstore(0);
  int cur = 0;
  for (int k0 = 0; k0 < K; k0 += 32) {
    __syncthreads();                 // buf[cur] visible to all waves
    const bool more = (k0 + 32 < K);
    if (more) gload(k0 + 32);        // in flight during compute below

    bf16x16 af[2], bfr[4];
#pragma unroll
    for (int mi = 0; mi < 2; ++mi)
      af[mi] = load_frag(Asm[cur] + (wr * 32 + mi * 16) * GLDT, GLDT);
#pragma unroll
    for (int j = 0; j < 4; ++j)
      bfr[j] = load_frag(Bsm[cur] + (wc * 64 + j * 16) * GLDT, GLDT);
#pragma unroll
    for (int mi = 0; mi < 2; ++mi)
#pragma unroll
      for (int j = 0; j < 4; ++j)
        acc[mi][j] = wmma_bf16(af[mi], bfr[j], acc[mi][j]);

    if (more) lstore(cur ^ 1);       // other buffer: no race with readers
    cur ^= 1;
  }

  const int n  = lane & 15;
  const int h2 = lane >> 4;
  float* cbase = C + n0 + wc * 64 + n;
  if (m0 + 128 <= M) {  // fast path: whole tile in range (uniform branch)
#pragma unroll
    for (int mi = 0; mi < 2; ++mi)
#pragma unroll
      for (int i = 0; i < 8; ++i) {
        const int grow = m0 + wr * 32 + mi * 16 + h2 * 8 + i;
        float* rowp = cbase + (size_t)grow * N;
#pragma unroll
        for (int j = 0; j < 4; ++j) rowp[j * 16] = acc[mi][j][i];
      }
  } else {  // partial tile (adapter GEMMs, M=10)
#pragma unroll
    for (int mi = 0; mi < 2; ++mi)
#pragma unroll
      for (int i = 0; i < 8; ++i) {
        const int grow = m0 + wr * 32 + mi * 16 + h2 * 8 + i;
        if (grow < M) {
          float* rowp = cbase + (size_t)grow * N;
#pragma unroll
          for (int j = 0; j < 4; ++j) rowp[j * 16] = acc[mi][j][i];
        }
      }
  }
}

// ---------------------------------------------------------------------------
// RoPE on q,k (interleaved pairs) -> bf16. d-fast mapping: coalesced in/out.
// ---------------------------------------------------------------------------
__global__ void rope_cvt(const float* __restrict__ qf, const float* __restrict__ kf,
                         const float* __restrict__ cb, const float* __restrict__ sb,
                         __bf16* __restrict__ qb, __bf16* __restrict__ kb) {
  int tid = blockIdx.x * blockDim.x + threadIdx.x;  // S*(D/2) threads
  int d2 = tid & (D_DIM / 2 - 1);  // pair index 0..2047
  int s  = tid >> 11;
  if (s >= S_LEN) return;
  int i = d2 & 63;  // rotary pair within head (HD/2=64)
  float c = cb[s * 64 + i], sn = sb[s * 64 + i];
  size_t base = (size_t)s * D_DIM + 2 * d2;
  float qr = qf[base], qi = qf[base + 1];
  qb[base]     = (__bf16)(qr * c - qi * sn);
  qb[base + 1] = (__bf16)(qr * sn + qi * c);
  float kr = kf[base], ki = kf[base + 1];
  kb[base]     = (__bf16)(kr * c - ki * sn);
  kb[base + 1] = (__bf16)(kr * sn + ki * c);
}

// V (S x D f32) -> per-head transposed bf16 vT[(h*128+hd)*S + s], LDS-tiled.
__global__ __launch_bounds__(256) void v_transpose(const float* __restrict__ vf,
                                                   __bf16* __restrict__ vT) {
  __shared__ float tile[32][33];
  const int s0 = blockIdx.x * 32;
  const int d0 = blockIdx.y * 32;
  const int tx = threadIdx.x & 31;
  const int ty = threadIdx.x >> 5;  // 0..7
#pragma unroll
  for (int r = 0; r < 4; ++r)
    tile[ty + r * 8][tx] = vf[(size_t)(s0 + ty + r * 8) * D_DIM + d0 + tx];
  __syncthreads();
#pragma unroll
  for (int r = 0; r < 4; ++r)
    vT[(size_t)(d0 + ty + r * 8) * S_LEN + s0 + tx] = (__bf16)tile[tx][ty + r * 8];
}

// adapter K padded to 16 rows, adapter V^T padded to 32 K-slots (zeros).
__global__ void adapter_cvt(const float* __restrict__ akf,
                            const float* __restrict__ avf,
                            __bf16* __restrict__ akb, __bf16* __restrict__ avT) {
  int tid = blockIdx.x * blockDim.x + threadIdx.x;  // 32*D threads
  int d = tid & (D_DIM - 1);
  int t = tid >> 12;
  if (t < 16)
    akb[(size_t)t * D_DIM + d] = (t < 10) ? (__bf16)akf[(size_t)t * D_DIM + d]
                                          : (__bf16)0.f;
  avT[(size_t)d * 32 + t] = (t < 10) ? (__bf16)avf[(size_t)t * D_DIM + d]
                                     : (__bf16)0.f;
}

// ---------------------------------------------------------------------------
// Flash attention: 1 head x 128 query rows per WG; wave owns 16 rows.
// Double-buffered K/V chunks staged with ASYNC global->LDS copies: chunk t+1
// streams in while WMMAs consume chunk t. Online softmax with causal mask +
// block-gate bias; padded adapter branch with its own softmax.
// ---------------------------------------------------------------------------
#define KLD 136  // 32x128 K chunk, padded row stride (bank-conflict-free)
#define VLD 40   // 128x32 V chunk, padded row stride
__global__ __launch_bounds__(256) void attn_wmma(
    const __bf16* __restrict__ qbf, const __bf16* __restrict__ kbf,
    const __bf16* __restrict__ vT, const __bf16* __restrict__ akb,
    const __bf16* __restrict__ avT, const float* __restrict__ gate,
    const float* __restrict__ gate2, const int* __restrict__ mfp,
    float* __restrict__ out) {
  __shared__ __align__(16) __bf16 Ksm[2][32 * KLD];
  __shared__ __align__(16) __bf16 Vsm[2][128 * VLD];
  __shared__ __align__(16) __bf16 psh[8 * 16 * 40];
  const int h    = blockIdx.y;
  const int wave = threadIdx.x >> 5;
  const int lane = threadIdx.x & 31;
  const int n    = lane & 15;
  const int h2   = lane >> 4;
  const int qblk = blockIdx.x * 128;
  const int qb0  = qblk + wave * 16;
  __bf16* pw = psh + wave * 16 * 40;
  const size_t hoff = (size_t)h * HD_DIM;

  auto stage = [&](int t0, int buf) {  // async: K 32x16 + V 128x4 granules
#pragma unroll
    for (int it = 0; it < 2; ++it) {
      int slot = it * 256 + threadIdx.x;
      int row = slot >> 4, g = slot & 15;
      async_copy16(kbf + (size_t)(t0 + row) * D_DIM + hoff + g * 8,
                   Ksm[buf] + row * KLD + g * 8);
    }
#pragma unroll
    for (int it = 0; it < 2; ++it) {
      int slot = it * 256 + threadIdx.x;
      int row = slot >> 2, g = slot & 3;
      async_copy16(vT + (hoff + row) * (size_t)S_LEN + t0 + g * 8,
                   Vsm[buf] + row * VLD + g * 8);
    }
  };

  bf16x16 qf[4];
#pragma unroll
  for (int kk = 0; kk < 4; ++kk)
    qf[kk] = load_frag(qbf + (size_t)qb0 * D_DIM + hoff + kk * 32, D_DIM);

  f32x8 O[8] = {};
  float mrun[8], lrun[8];
#pragma unroll
  for (int i = 0; i < 8; ++i) { mrun[i] = -__builtin_inff(); lrun[i] = 0.f; }

  const float scale = 0.08838834764831845f;  // 1/sqrt(128)
  const int   mf = mfp[0];
  const float g2 = gate2[h];
  const int   t_hi  = qblk + 127;  // block-uniform causal bound
  const int   sq_hi = qb0 + 15;    // wave causal bound

  stage(0, 0);
  async_join();
  __syncthreads();
  int cur = 0;
  for (int t0 = 0; t0 <= t_hi; t0 += 32) {
    if (t0 + 32 <= t_hi) stage(t0 + 32, cur ^ 1);  // overlap with compute

    if (t0 <= sq_hi) {  // skip fully-masked chunks for this wave's rows
      f32x8 s0 = {}, s1 = {};
#pragma unroll
      for (int kk = 0; kk < 4; ++kk) {
        bf16x16 b0 = load_frag(Ksm[cur] + kk * 32, KLD);
        bf16x16 b1 = load_frag(Ksm[cur] + 16 * KLD + kk * 32, KLD);
        s0 = wmma_bf16(qf[kk], b0, s0);
        s1 = wmma_bf16(qf[kk], b1, s1);
      }
#pragma unroll
      for (int i = 0; i < 8; ++i) {
        const int sq = qb0 + h2 * 8 + i;
        const int ta = t0 + n, tb = t0 + 16 + n;
        float va = s0[i] * scale + ((ta > sq) ? -1.0e9f : 0.f) +
                   (((sq >= mf) && (ta < mf)) ? g2 : 0.f);
        float vb = s1[i] * scale + ((tb > sq) ? -1.0e9f : 0.f) +
                   (((sq >= mf) && (tb < mf)) ? g2 : 0.f);
        float mnew  = fmaxf(mrun[i], rmax16(fmaxf(va, vb)));
        float alpha = __expf(mrun[i] - mnew);
        float pa = __expf(va - mnew), pb = __expf(vb - mnew);
        lrun[i] = lrun[i] * alpha + rsum16(pa + pb);
        mrun[i] = mnew;
#pragma unroll
        for (int j = 0; j < 8; ++j) O[j][i] *= alpha;
        const int row = h2 * 8 + i;
        pw[row * 40 + n]      = (__bf16)pa;
        pw[row * 40 + 16 + n] = (__bf16)pb;
      }
      bf16x16 pf = load_frag(pw, 40);
#pragma unroll
      for (int j = 0; j < 8; ++j)
        O[j] = wmma_bf16(pf, load_frag(Vsm[cur] + (j * 16) * VLD, VLD), O[j]);
    }

    async_join();      // this wave's copies into buf^1 have landed
    __syncthreads();   // everyone's have; buf^1 becomes readable
    cur ^= 1;
  }
#pragma unroll
  for (int i = 0; i < 8; ++i) {
    float inv = 1.0f / lrun[i];
#pragma unroll
    for (int j = 0; j < 8; ++j) O[j][i] *= inv;
  }
  {  // adapter branch: separate softmax over 10 keys, scaled by tanh(gate)
    f32x8 sa = {};
#pragma unroll
    for (int kk = 0; kk < 4; ++kk)
      sa = wmma_bf16(qf[kk], load_frag(akb + hoff + kk * 32, D_DIM), sa);
    const float tg = tanhf(gate[h]);
#pragma unroll
    for (int i = 0; i < 8; ++i) {
      float v = (n < 10) ? sa[i] * scale : -1.0e30f;
      float rm = rmax16(v);
      float p  = __expf(v - rm);
      float pr = (p / rsum16(p)) * tg;
      const int row = h2 * 8 + i;
      pw[row * 40 + n]      = (__bf16)pr;
      pw[row * 40 + 16 + n] = (__bf16)0.f;
    }
    bf16x16 pf = load_frag(pw, 40);
#pragma unroll
    for (int j = 0; j < 8; ++j)
      O[j] = wmma_bf16(pf, load_frag(avT + (hoff + j * 16) * 32, 32), O[j]);
  }
  float* obase = out + hoff + n;
#pragma unroll
  for (int i = 0; i < 8; ++i) {
    const int sq = qb0 + h2 * 8 + i;
    float* rowp = obase + (size_t)sq * D_DIM;
#pragma unroll
    for (int j = 0; j < 8; ++j) rowp[j * 16] = O[j][i];
  }
}

extern "C" void kernel_launch(void* const* d_in, const int* in_sizes, int n_in,
                              void* d_out, int out_size, void* d_ws, size_t ws_size,
                              hipStream_t stream) {
  (void)in_sizes; (void)n_in; (void)out_size; (void)ws_size;
  const float* x   = (const float*)d_in[0];
  const float* ad  = (const float*)d_in[1];
  const float* wq  = (const float*)d_in[2];
  const float* wk  = (const float*)d_in[3];
  const float* wv  = (const float*)d_in[4];
  const float* wo  = (const float*)d_in[5];
  const float* gt  = (const float*)d_in[6];
  const float* gt2 = (const float*)d_in[7];
  const float* fc  = (const float*)d_in[8];
  const float* fs  = (const float*)d_in[9];
  const int*   mf  = (const int*)d_in[11];
  float* out = (float*)d_out;

  char* ws = (char*)d_ws;
  size_t off = 0;
  auto carve = [&](size_t bytes) -> char* {
    char* p = ws + off;
    off += (bytes + 255) & ~(size_t)255;
    return p;
  };
  const size_t SD = (size_t)S_LEN * D_DIM;
  float*  qf32 = (float*)carve(SD * 4);
  float*  kf32 = (float*)carve(SD * 4);
  float*  vf32 = (float*)carve(SD * 4);
  float*  akf  = (float*)carve((size_t)16 * D_DIM * 4);
  float*  avf  = (float*)carve((size_t)16 * D_DIM * 4);
  __bf16* qbf  = (__bf16*)carve(SD * 2);
  __bf16* kbf  = (__bf16*)carve(SD * 2);
  __bf16* vT   = (__bf16*)carve(SD * 2);
  __bf16* akb  = (__bf16*)carve((size_t)16 * D_DIM * 2);
  __bf16* avT  = (__bf16*)carve((size_t)D_DIM * 32 * 2);
  float*  attn = qf32;  // qf32 is dead after rope_cvt -> reuse for attn output

  dim3 blk(256);
  dim3 gP(D_DIM / 128, S_LEN / 128);  // (32,16)
  gemm_wmma<<<gP, blk, 0, stream>>>(x, wq, qf32, S_LEN, D_DIM, D_DIM);
  gemm_wmma<<<gP, blk, 0, stream>>>(x, wk, kf32, S_LEN, D_DIM, D_DIM);
  gemm_wmma<<<gP, blk, 0, stream>>>(x, wv, vf32, S_LEN, D_DIM, D_DIM);
  dim3 gA(D_DIM / 128, 1);
  gemm_wmma<<<gA, blk, 0, stream>>>(ad, wk, akf, 10, D_DIM, D_DIM);
  gemm_wmma<<<gA, blk, 0, stream>>>(ad, wv, avf, 10, D_DIM, D_DIM);

  rope_cvt<<<(S_LEN * (D_DIM / 2)) / 256, blk, 0, stream>>>(qf32, kf32, fc, fs,
                                                            qbf, kbf);
  dim3 gV(S_LEN / 32, D_DIM / 32);  // (64,128)
  v_transpose<<<gV, blk, 0, stream>>>(vf32, vT);
  adapter_cvt<<<(32 * D_DIM) / 256, blk, 0, stream>>>(akf, avf, akb, avT);

  dim3 gAt(S_LEN / 128, H_NUM);  // (16,32)
  attn_wmma<<<gAt, blk, 0, stream>>>(qbf, kbf, vT, akb, avT, gt, gt2, mf, attn);

  gemm_wmma<<<gP, blk, 0, stream>>>(attn, wo, out, S_LEN, D_DIM, D_DIM);
}
